// MambaLayer_45045617000489
// MI455X (gfx1250) — compile-verified, matrix operand order checked by reading, not run on hardware
//
#include <hip/hip_runtime.h>
#include <stdint.h>

// ---------------- problem constants ----------------
constexpr int NB     = 8;
constexpr int LTOK   = 1024;         // H*W
constexpr int NTOK   = NB * LTOK;    // 8192
constexpr int CINC   = 512;
constexpr int DINNER = 1024;
constexpr int DSTATE = 64;
constexpr int NH     = 16;
constexpr int HD     = 64;
constexpr int DXBC   = DINNER + 2 * DSTATE;            // 1152
constexpr int DPROJ  = 2 * DINNER + 2 * DSTATE + NH;   // 2192
constexpr float EPSF = 1e-5f;

// ---------------- types ----------------
typedef __attribute__((ext_vector_type(16))) __bf16 v16bf;
typedef __attribute__((ext_vector_type(8)))  float  v8f;
typedef __attribute__((ext_vector_type(4)))  unsigned int uv4;
typedef __attribute__((ext_vector_type(4)))  unsigned int u32x4;
typedef __attribute__((ext_vector_type(8)))  int i32x8;
typedef __attribute__((ext_vector_type(4)))  int i32x4;

__device__ __forceinline__ unsigned short f2bf(float f) {
  union { float f; unsigned u; } c; c.f = f;
  return (unsigned short)((c.u + 0x7fffu + ((c.u >> 16) & 1u)) >> 16);  // RNE
}

// ---------------- TDM: load 'nrows' consecutive rows (row-major, K bf16 each)
// of a (Ntotal x K) bf16 matrix into LDS offset 0. Wave 0 issues the DMA.
// The LDS base pointer is passed *into* the asm barrier: that escapes the
// internal LDS global, so the "memory" clobber is assumed to write it and the
// ds_load consumers cannot be folded to undef. ----
__device__ __forceinline__ void tdm_load_rows_bf16(unsigned short* lds,
                                                   const unsigned short* gsrc,
                                                   unsigned K, unsigned nrows,
                                                   unsigned Ntotal) {
  if ((threadIdx.x >> 5) == 0) {     // one wave issues the DMA
    uint64_t ga = (uint64_t)(uintptr_t)gsrc;
    u32x4 g0;
    g0[0] = 1u;                                       // count=1, user desc
    g0[1] = 0u;                                       // lds_addr = 0
    g0[2] = (unsigned)(ga & 0xffffffffull);           // global_addr lo
    g0[3] = (unsigned)((ga >> 32) & 0x01ffffffu) | (2u << 30); // hi | type=2
    i32x8 g1;
    g1[0] = (int)(1u << 16);                          // data_size = 2 bytes
    g1[1] = (int)((K & 0xffffu) << 16);               // tensor_dim0 lo16
    g1[2] = (int)((K >> 16) | ((Ntotal & 0xffffu) << 16)); // dim0 hi | dim1 lo
    g1[3] = (int)((Ntotal >> 16) | ((K & 0xffffu) << 16)); // dim1 hi | tile_dim0=K
    g1[4] = (int)(nrows & 0xffffu);                   // tile_dim1
    g1[5] = (int)K;                                   // tensor_dim0_stride lo32
    g1[6] = 0;
    g1[7] = 0;
    i32x4 z4 = {0, 0, 0, 0};
    i32x8 z8 = {0, 0, 0, 0, 0, 0, 0, 0};
    __builtin_amdgcn_tensor_load_to_lds(g0, g1, z4, z4, z8, 0);
    __builtin_amdgcn_s_wait_tensorcnt(0);
  }
  // Escape the LDS address + full memory clobber: the TDM write is now
  // something the compiler must assume happened.
  void* p = (void*)lds;
  asm volatile("" : "+v"(p) : : "memory");
}

// ---------------- fp32 -> bf16 conversion ----------------
__global__ void __launch_bounds__(256) k_cvt_bf16(const float* __restrict__ s,
                                                  unsigned short* __restrict__ d, int n) {
  int i = blockIdx.x * 256 + threadIdx.x;
  if (i < n) d[i] = f2bf(s[i]);
}

// ---------------- fused W_f = conv2d_w(512x512) @ out_proj_w(512x1024) -> bf16 ----
__global__ void __launch_bounds__(256) k_fuse_w(const float* __restrict__ cw,
                                                const float* __restrict__ ow,
                                                unsigned short* __restrict__ wf) {
  int d = blockIdx.x * 256 + threadIdx.x;   // 0..1023
  int o = blockIdx.y;                       // 0..511
  float acc = 0.f;
  const float* cr = cw + (size_t)o * CINC;
  for (int c = 0; c < CINC; ++c) acc += cr[c] * ow[(size_t)c * DINNER + d];
  wf[(size_t)o * DINNER + d] = f2bf(acc);
}

// ---------------- LayerNorm over channels per token; emit bf16 ----------------
__global__ void __launch_bounds__(256) k_layernorm(const float* __restrict__ x,
                                                   const float* __restrict__ lw,
                                                   const float* __restrict__ lb,
                                                   unsigned short* __restrict__ xn) {
  int t = blockIdx.x;                 // 0..8191
  int bb = t >> 10, l = t & 1023;
  const float* xp = x + ((size_t)bb * CINC) * LTOK + l;   // + c*1024
  int c0 = threadIdx.x, c1 = threadIdx.x + 256;
  float v0 = xp[(size_t)c0 * LTOK];
  float v1 = xp[(size_t)c1 * LTOK];
  float s = v0 + v1, s2 = v0 * v0 + v1 * v1;
  for (int o = 16; o; o >>= 1) { s += __shfl_xor(s, o, 32); s2 += __shfl_xor(s2, o, 32); }
  __shared__ float red[16];
  int w = threadIdx.x >> 5;
  if ((threadIdx.x & 31) == 0) { red[w] = s; red[w + 8] = s2; }
  __syncthreads();
  if (threadIdx.x == 0) {
    float a = 0.f, b2 = 0.f;
    for (int i = 0; i < 8; ++i) { a += red[i]; b2 += red[i + 8]; }
    red[0] = a; red[8] = b2;
  }
  __syncthreads();
  float mean = red[0] * (1.0f / CINC);
  float var  = red[8] * (1.0f / CINC) - mean * mean;
  float rstd = rsqrtf(var + EPSF);
  xn[(size_t)t * CINC + c0] = f2bf((v0 - mean) * rstd * lw[c0] + lb[c0]);
  xn[(size_t)t * CINC + c1] = f2bf((v1 - mean) * rstd * lw[c1] + lb[c1]);
}

// ---------------- fragment loader: 8 bf16 @p, 8 bf16 @p+16 (ISA 16-bit layout) ----
__device__ __forceinline__ v16bf load_frag(const unsigned short* p) {
  union { uv4 u[2]; v16bf v; } t;
  t.u[0] = *(const uv4*)(const void*)(p);
  t.u[1] = *(const uv4*)(const void*)(p + 16);
  return t.v;
}

// ---------------- in_proj GEMM: zx(8192x2192) = xn(8192x512,bf16) @ Wb^T -------
// block = 8 waves; wave w -> rows m0+16w; block cols n0..n0+15; TDM panel in LDS.
__global__ void __launch_bounds__(256) k_inproj_gemm(const unsigned short* __restrict__ A,
                                                     const unsigned short* __restrict__ Wb,
                                                     float* __restrict__ C) {
  __shared__ unsigned short sW[16 * CINC];          // 16 KB weight panel
  const int n0 = blockIdx.y * 16;
  const int m0 = blockIdx.x * 128 + (int)(threadIdx.x >> 5) * 16;
  tdm_load_rows_bf16(sW, Wb + (size_t)n0 * CINC, CINC, 16, DPROJ);
  __syncthreads();
  const int lane = threadIdx.x & 31;
  const int half = lane >> 4;
  const int l16  = lane & 15;
  v8f acc = {};
  const unsigned short* arow = A  + (size_t)(m0 + l16) * CINC + half * 8;
  const unsigned short* brow = sW + (size_t)l16 * CINC + half * 8;
#pragma unroll 4
  for (int k = 0; k < CINC; k += 32) {
    v16bf af = load_frag(arow + k);
    v16bf bf = load_frag(brow + k);
    acc = __builtin_amdgcn_wmma_f32_16x16x32_bf16(false, af, false, bf,
                                                  (short)0, acc, false, false);
  }
  const int col = n0 + l16;
#pragma unroll
  for (int v = 0; v < 8; ++v) {
    int row = m0 + half * 8 + v;
    C[(size_t)row * DPROJ + col] = acc[v];
  }
}

// ---------------- depthwise causal conv1d + SiLU ----------------
__global__ void __launch_bounds__(256) k_conv1d(const float* __restrict__ zx,
                                                const float* __restrict__ w,
                                                const float* __restrict__ bias,
                                                float* __restrict__ xbc) {
  int d  = blockIdx.x;          // 0..1151
  int bb = blockIdx.y;          // 0..7
  float w0 = w[d * 4 + 0], w1 = w[d * 4 + 1], w2 = w[d * 4 + 2], w3 = w[d * 4 + 3];
  float bs = bias[d];
  const float* src = zx + ((size_t)bb * LTOK) * DPROJ + DINNER + d;
  float* dst = xbc + ((size_t)bb * LTOK) * DXBC + d;
  for (int l = threadIdx.x; l < LTOK; l += 256) {
    float a0 = (l >= 3) ? src[(size_t)(l - 3) * DPROJ] : 0.f;
    float a1 = (l >= 2) ? src[(size_t)(l - 2) * DPROJ] : 0.f;
    float a2 = (l >= 1) ? src[(size_t)(l - 1) * DPROJ] : 0.f;
    float a3 = src[(size_t)l * DPROJ];
    float r = a0 * w0 + a1 * w1 + a2 * w2 + a3 * w3 + bs;
    r = r / (1.f + __expf(-r));                   // SiLU
    dst[(size_t)l * DXBC] = r;
  }
}

// ---------------- dt softplus ----------------
__global__ void __launch_bounds__(256) k_dt(const float* __restrict__ zx,
                                            const float* __restrict__ dtb,
                                            float* __restrict__ dt) {
  int i = blockIdx.x * 256 + threadIdx.x;
  if (i >= NTOK * NH) return;
  int t = i >> 4, h = i & 15;
  float v = zx[(size_t)t * DPROJ + (2 * DINNER + 2 * DSTATE) + h] + dtb[h];
  dt[i] = (v > 20.f) ? v : log1pf(__expf(v));
}

// ---------------- selective scan: 1 block per (batch, head) ----------------
__global__ void __launch_bounds__(256) k_scan(const float* __restrict__ xbc,
                                              const float* __restrict__ dtv,
                                              const float* __restrict__ A_log,
                                              float* __restrict__ ys) {
  int bb = blockIdx.x >> 4;
  int hh = blockIdx.x & 15;
  float Ac = -__expf(A_log[hh]);
  __shared__ float sx[HD], sB[DSTATE], sC[DSTATE], sdt[1];
  int tid = threadIdx.x;
  int p = tid >> 2, q = tid & 3, s0 = q * 16;
  float hreg[16];
#pragma unroll
  for (int j = 0; j < 16; ++j) hreg[j] = 0.f;
  const float* base = xbc + ((size_t)bb * LTOK) * DXBC;
  const float* dtb  = dtv + ((size_t)bb * LTOK) * NH + hh;
  float* yb = ys + ((size_t)bb * LTOK) * DINNER + hh * HD;
  for (int l = 0; l < LTOK; ++l) {
    const float* row = base + (size_t)l * DXBC;
    if (tid < 64)       sx[tid]       = row[hh * HD + tid];
    else if (tid < 128) sB[tid - 64]  = row[DINNER + (tid - 64)];
    else if (tid < 192) sC[tid - 128] = row[DINNER + DSTATE + (tid - 128)];
    else if (tid == 192) sdt[0] = dtb[(size_t)l * NH];
    __syncthreads();
    float dt = sdt[0];
    float dA = __expf(dt * Ac);
    float dx = dt * sx[p];
    float acc = 0.f;
#pragma unroll
    for (int j = 0; j < 16; ++j) {
      hreg[j] = hreg[j] * dA + dx * sB[s0 + j];
      acc += hreg[j] * sC[s0 + j];
    }
    acc += __shfl_xor(acc, 1, 32);      // reduce over state quads (wave32)
    acc += __shfl_xor(acc, 2, 32);
    if (q == 0) yb[(size_t)l * DINNER + p] = acc;
    __syncthreads();
  }
}

// ---------------- y = (ys + D*x) * silu(z); RMSNorm; -> bf16 ----------------
__global__ void __launch_bounds__(256) k_gate_rms(const float* __restrict__ zx,
                                                  const float* __restrict__ xbc,
                                                  const float* __restrict__ ys,
                                                  const float* __restrict__ Dp,
                                                  const float* __restrict__ rw,
                                                  unsigned short* __restrict__ ybf) {
  int t = blockIdx.x;
  const float* zr = zx  + (size_t)t * DPROJ;
  const float* xr = xbc + (size_t)t * DXBC;
  const float* yr = ys  + (size_t)t * DINNER;
  float vals[4]; float ss = 0.f;
#pragma unroll
  for (int i = 0; i < 4; ++i) {
    int d = threadIdx.x + i * 256;
    float yv = yr[d] + Dp[d >> 6] * xr[d];
    float z  = zr[d];
    float g  = z / (1.f + __expf(-z));
    float r  = yv * g;
    vals[i] = r; ss += r * r;
  }
  for (int o = 16; o; o >>= 1) ss += __shfl_xor(ss, o, 32);
  __shared__ float red[8];
  int w = threadIdx.x >> 5;
  if ((threadIdx.x & 31) == 0) red[w] = ss;
  __syncthreads();
  if (threadIdx.x == 0) {
    float a = 0.f;
    for (int i = 0; i < 8; ++i) a += red[i];
    red[0] = a;
  }
  __syncthreads();
  float scale = rsqrtf(red[0] * (1.0f / DINNER) + EPSF);
#pragma unroll
  for (int i = 0; i < 4; ++i) {
    int d = threadIdx.x + i * 256;
    ybf[(size_t)t * DINNER + d] = f2bf(vals[i] * scale * rw[d]);
  }
}

// ---------------- fused out_proj + 1x1 conv GEMM + BN + ReLU ----------------
__global__ void __launch_bounds__(256) k_out_gemm(const unsigned short* __restrict__ A,
                                                  const unsigned short* __restrict__ Wf,
                                                  const float* __restrict__ cb,
                                                  const float* __restrict__ bg,
                                                  const float* __restrict__ bbx,
                                                  const float* __restrict__ bm,
                                                  const float* __restrict__ bv,
                                                  float* __restrict__ out) {
  __shared__ unsigned short sW[16 * DINNER];        // 32 KB weight panel
  const int n0 = blockIdx.y * 16;
  const int m0 = blockIdx.x * 128 + (int)(threadIdx.x >> 5) * 16;
  tdm_load_rows_bf16(sW, Wf + (size_t)n0 * DINNER, DINNER, 16, CINC);
  __syncthreads();
  const int lane = threadIdx.x & 31;
  const int half = lane >> 4;
  const int l16  = lane & 15;
  v8f acc = {};
  const unsigned short* arow = A  + (size_t)(m0 + l16) * DINNER + half * 8;
  const unsigned short* brow = sW + (size_t)l16 * DINNER + half * 8;
#pragma unroll 4
  for (int k = 0; k < DINNER; k += 32) {
    v16bf af = load_frag(arow + k);
    v16bf bf = load_frag(brow + k);
    acc = __builtin_amdgcn_wmma_f32_16x16x32_bf16(false, af, false, bf,
                                                  (short)0, acc, false, false);
  }
  const int c  = n0 + l16;
  float sc = bg[c] * rsqrtf(bv[c] + EPSF);
  float sh = bbx[c] + (cb[c] - bm[c]) * sc;
#pragma unroll
  for (int v = 0; v < 8; ++v) {
    int m = m0 + half * 8 + v;
    int b_ = m >> 10, l = m & 1023;
    float r = fmaxf(acc[v] * sc + sh, 0.f);
    out[((size_t)(b_ * CINC + c)) * LTOK + l] = r;   // (B, C_OUT, H, W)
  }
}

// ---------------- launcher ----------------
extern "C" void kernel_launch(void* const* d_in, const int* in_sizes, int n_in,
                              void* d_out, int out_size, void* d_ws, size_t ws_size,
                              hipStream_t stream) {
  const float* x       = (const float*)d_in[0];
  const float* ln_w    = (const float*)d_in[1];
  const float* ln_b    = (const float*)d_in[2];
  const float* inpw    = (const float*)d_in[3];
  const float* c1w     = (const float*)d_in[4];
  const float* c1b     = (const float*)d_in[5];
  const float* dtbias  = (const float*)d_in[6];
  const float* A_log   = (const float*)d_in[7];
  const float* Dp      = (const float*)d_in[8];
  const float* rms_w   = (const float*)d_in[9];
  const float* opw     = (const float*)d_in[10];
  const float* c2w     = (const float*)d_in[11];
  const float* c2b     = (const float*)d_in[12];
  const float* bn_g    = (const float*)d_in[13];
  const float* bn_b    = (const float*)d_in[14];
  const float* bn_mean = (const float*)d_in[15];
  const float* bn_var  = (const float*)d_in[16];
  float* out = (float*)d_out;

  size_t off = 0;
  char* ws = (char*)d_ws;
  auto alloc = [&](size_t bytes) -> void* {
    void* p = ws + off;
    off += (bytes + 255) & ~(size_t)255;
    return p;
  };
  unsigned short* Wb  = (unsigned short*)alloc((size_t)DPROJ * CINC * 2);
  unsigned short* Wf  = (unsigned short*)alloc((size_t)CINC * DINNER * 2);
  unsigned short* xn  = (unsigned short*)alloc((size_t)NTOK * CINC * 2);
  float* zx   = (float*)alloc((size_t)NTOK * DPROJ * 4);
  float* xbc  = (float*)alloc((size_t)NTOK * DXBC * 4);
  float* dtv  = (float*)alloc((size_t)NTOK * NH * 4);
  float* ys   = (float*)alloc((size_t)NTOK * DINNER * 4);
  unsigned short* ybf = (unsigned short*)alloc((size_t)NTOK * DINNER * 2);

  (void)in_sizes; (void)n_in; (void)out_size; (void)ws_size;

  // 1. weight prep
  int nInW = DPROJ * CINC;
  k_cvt_bf16<<<(nInW + 255) / 256, 256, 0, stream>>>(inpw, Wb, nInW);
  k_fuse_w<<<dim3(DINNER / 256, CINC), 256, 0, stream>>>(c2w, opw, Wf);
  // 2. layernorm -> bf16 activations
  k_layernorm<<<NTOK, 256, 0, stream>>>(x, ln_w, ln_b, xn);
  // 3. in_proj GEMM (WMMA bf16, TDM-staged weights)
  k_inproj_gemm<<<dim3(NTOK / 128, DPROJ / 16), 256, 0, stream>>>(xn, Wb, zx);
  // 4. depthwise conv1d + SiLU
  k_conv1d<<<dim3(DXBC, NB), 256, 0, stream>>>(zx, c1w, c1b, xbc);
  // 5. dt softplus
  k_dt<<<(NTOK * NH + 255) / 256, 256, 0, stream>>>(zx, dtbias, dtv);
  // 6. selective scan
  k_scan<<<NB * NH, 256, 0, stream>>>(xbc, dtv, A_log, ys);
  // 7. gate + RMSNorm -> bf16
  k_gate_rms<<<NTOK, 256, 0, stream>>>(zx, xbc, ys, Dp, rms_w, ybf);
  // 8. fused out_proj + conv2d + BN + ReLU (WMMA bf16, TDM-staged weights)
  k_out_gemm<<<dim3(NTOK / 128, CINC / 16), 256, 0, stream>>>(
      ybf, Wf, c2b, bn_g, bn_b, bn_mean, bn_var, out);
}
